// SlidingWindowAttention_68719476736026
// MI455X (gfx1250) — matrix operand order repaired and checked by reading, not compiled, not run on hardware
//
#include <hip/hip_runtime.h>

typedef __attribute__((ext_vector_type(16))) _Float16 v16h;
typedef __attribute__((ext_vector_type(8)))  float    v8f;

namespace {
constexpr int KW   = 7;
constexpr int NH   = 8;
constexpr int HW   = 56;
constexpr int D    = 32;
constexpr int HW2  = HW * HW;        // 3136
constexpr int UNW  = 22;             // key-union width (columns)
constexpr int UNK  = KW * UNW;       // 154 union keys
constexpr int NT   = 10;             // S tiles of 16 keys (padded to 160)
constexpr int SLD  = 160;
}

__global__ __launch_bounds__(128)
void natten77_wmma_kernel(const float* __restrict__ x,
                          const float* __restrict__ rpb,
                          float* __restrict__ out)
{
  __shared__ float    Ssm[4][16][SLD];   // logits / scratch (per wave)
  __shared__ _Float16 Psm[4][16][SLD];   // unnormalized probs (f16)
  __shared__ float    Isum[4][16];       // 1/rowsum

  const int lane = threadIdx.x & 31;
  const int w    = threadIdx.x >> 5;
  const int i    = blockIdx.x % HW;
  const int head = (blockIdx.x / HW) % NH;
  const int b    = blockIdx.x / (HW * NH);

  const int j0  = (w == 3) ? 40 : w * 16;          // query-column tile
  const int ri0 = min(max(i - 3, 0), HW - KW);     // window row start (shared by row)
  const int pi  = (KW - 1) - (i - ri0);            // rpb row base
  const int c0  = min(min(max(j0 - 3, 0), HW - KW), HW - UNW); // union col start

  const int xb = b * (3 * 256 * HW2);
  const float* qp = x + xb + (head * D) * HW2 + i * HW;        // q[., dch, i, j]
  const float* kp = x + xb + (256 + head * D) * HW2;           // k[., dch, r, c]
  const float* vp = x + xb + (512 + head * D) * HW2;           // v[., dch, r, c]

  const int ml   = lane & 15;
  const int hi8  = (lane & 16) ? 8  : 0;
  const int hi16 = (lane & 16) ? 16 : 0;
  const float scale = 0.17677669529663687f;  // 1/sqrt(32)

  union F16x16 { v16h v; _Float16 e[16]; };

  // ---- Q fragment: A-matrix 16x32 f16, pre-scaled ----
  F16x16 qa;
  #pragma unroll
  for (int r = 0; r < 8; ++r) {
    const int ka = 2 * (r & 3) + hi8 + ((r & 4) ? 16 : 0);
    qa.e[2*r]   = (_Float16)(qp[ka * HW2 + j0 + ml] * scale);
    qa.e[2*r+1] = (_Float16)(qp[(ka + 1) * HW2 + j0 + ml] * scale);
  }

  // ---- S = Q @ K^T over the 7x22 key union (10 N-tiles) ----
  for (int t = 0; t < NT; ++t) {
    int kk = t * 16 + ml; if (kk > UNK - 1) kk = UNK - 1;   // pad keys clamp
    const int kr = kk / UNW;
    const int kc = kk - kr * UNW;
    const float* kcol = kp + (ri0 + kr) * HW + (c0 + kc);
    F16x16 kb;
    #pragma unroll
    for (int r = 0; r < 8; ++r) {
      const int kbk = 2 * (r & 3) + hi8 + ((r & 4) ? 16 : 0);          // B-matrix K striping
      kb.e[2*r]   = (_Float16)kcol[kbk * HW2];
      kb.e[2*r+1] = (_Float16)kcol[(kbk + 1) * HW2];
    }
    v8f acc = {};
    acc = __builtin_amdgcn_wmma_f32_16x16x32_f16(false, qa.v, false, kb.v,
                                                 (short)0, acc, false, false);
    #pragma unroll
    for (int r = 0; r < 8; ++r)
      Ssm[w][r + hi8][t * 16 + ml] = acc[r];
  }

  // ---- bias + softmax over valid 49 keys (one query row per lane 0..15) ----
  if (lane < 16) {
    const int m  = lane;
    const int j  = j0 + m;
    const int nj = min(max(j - 3, 0), HW - KW);
    const int pj = (KW - 1) - (j - nj);
    const float* rb = rpb + head * 169 + pi * 13 + pj;
    float mx = -3.0e38f;
    for (int n = 0; n < SLD; ++n) {
      float lg = -3.0e38f;
      if (n < UNK) {
        const int kr = n / UNW;
        const int kj = (c0 + (n - kr * UNW)) - nj;
        if ((unsigned)kj < (unsigned)KW) {
          lg = Ssm[w][m][n] + rb[kr * 13 + kj];
          mx = fmaxf(mx, lg);
        }
      }
      Ssm[w][m][n] = lg;
    }
    float sum = 0.0f;
    for (int n = 0; n < SLD; ++n) {
      const float p = __expf(Ssm[w][m][n] - mx);   // invalid/pad -> 0
      sum += p;
      Psm[w][m][n] = (_Float16)p;
    }
    Isum[w][m] = 1.0f / sum;
  }

  // ---- O = P @ V (K = 160 in 5 chunks of 32; two 16-wide d tiles) ----
  v8f o0 = {}, o1 = {};
  #pragma unroll
  for (int c = 0; c < 5; ++c) {
    F16x16 pa, vb0, vb1;
    #pragma unroll
    for (int r = 0; r < 8; ++r) {
      const int ka = 32 * c + 2 * (r & 3) + hi8 + ((r & 4) ? 16 : 0);
      pa.e[2*r]   = Psm[w][ml][ka];
      pa.e[2*r+1] = Psm[w][ml][ka + 1];
    }
    #pragma unroll
    for (int r = 0; r < 8; ++r) {
      const int kbk = 32 * c + 2 * (r & 3) + hi8 + ((r & 4) ? 16 : 0);
      const int kk0 = min(kbk,     UNK - 1);
      const int kk1 = min(kbk + 1, UNK - 1);
      const int kr0 = kk0 / UNW, kc0 = kk0 - kr0 * UNW;
      const int kr1 = kk1 / UNW, kc1 = kk1 - kr1 * UNW;
      const int a0 = (ri0 + kr0) * HW + (c0 + kc0);
      const int a1 = (ri0 + kr1) * HW + (c0 + kc1);
      vb0.e[2*r]   = (_Float16)vp[ml * HW2 + a0];
      vb0.e[2*r+1] = (_Float16)vp[ml * HW2 + a1];
      vb1.e[2*r]   = (_Float16)vp[(16 + ml) * HW2 + a0];
      vb1.e[2*r+1] = (_Float16)vp[(16 + ml) * HW2 + a1];
    }
    o0 = __builtin_amdgcn_wmma_f32_16x16x32_f16(false, pa.v, false, vb0.v,
                                                (short)0, o0, false, false);
    o1 = __builtin_amdgcn_wmma_f32_16x16x32_f16(false, pa.v, false, vb1.v,
                                                (short)0, o1, false, false);
  }

  // ---- normalize (fold 1/rowsum) and store; wave3 writes only j>=48 ----
  const int ob = b * (256 * HW2) + (head * D) * HW2 + i * HW;
  #pragma unroll
  for (int r = 0; r < 8; ++r) {
    const int m = r + hi8;
    if ((w != 3) || (m >= 8)) {
      const float is = Isum[w][m];
      out[ob + ml * HW2 + j0 + m]        = o0[r] * is;
      out[ob + (16 + ml) * HW2 + j0 + m] = o1[r] * is;
    }
  }
}

extern "C" void kernel_launch(void* const* d_in, const int* in_sizes, int n_in,
                              void* d_out, int out_size, void* d_ws, size_t ws_size,
                              hipStream_t stream) {
  (void)in_sizes; (void)n_in; (void)d_ws; (void)ws_size; (void)out_size;
  const float* x   = (const float*)d_in[0];
  const float* rpb = (const float*)d_in[1];
  float* out       = (float*)d_out;
  dim3 grid(2 * NH * HW);   // (b, head, row) = 896 blocks
  dim3 block(128);          // 4 waves, one 16-query j-tile each
  natten77_wmma_kernel<<<grid, block, 0, stream>>>(x, rpb, out);
}
